// CrossAttnGPTBigCodeBlock_77695958384909
// MI455X (gfx1250) — matrix operand order repaired and checked by reading, not compiled, not run on hardware
//
#include <hip/hip_runtime.h>
#include <hip/hip_bf16.h>
#include <math.h>

// Problem constants (from reference)
#define D_MODEL 2048
#define NHEAD 16
#define HEADDIM 128
#define INNER_DIM 8192
#define BATCH 4
#define LQ_ 1024
#define LK_ 2048

typedef __bf16 v16bf __attribute__((ext_vector_type(16)));
typedef float v8f __attribute__((ext_vector_type(8)));
typedef int v4i __attribute__((ext_vector_type(4)));

union Frag { uint4 q[2]; v16bf v; };

// ---- CDNA5 async global->LDS copy path (ASYNCcnt), guarded ---------------
#if defined(__AMDGCN__) && __has_builtin(__builtin_amdgcn_global_load_async_to_lds_b128)
#define HAS_ASYNC_LDS 1
#else
#define HAS_ASYNC_LDS 0
#endif

#if HAS_ASYNC_LDS
__device__ __forceinline__ void async_copy_b128(const void* gsrc, void* ldst) {
  // Builtin signature (from clang diagnostic): pointers to 16B int vectors.
  // LDS flat address truncates to the LDS byte offset (ISA 10.2 aperture rule).
  __builtin_amdgcn_global_load_async_to_lds_b128(
      (__attribute__((address_space(1))) v4i*)(unsigned long long)gsrc,
      (__attribute__((address_space(3))) v4i*)(unsigned long long)(unsigned)(
          unsigned long long)ldst,
      0, 0);
}
__device__ __forceinline__ void wait_async_all() {
#if __has_builtin(__builtin_amdgcn_s_wait_asynccnt)
  __builtin_amdgcn_s_wait_asynccnt(0);
#else
  asm volatile("s_wait_asynccnt 0" ::: "memory");
#endif
}
#endif

__device__ __forceinline__ unsigned short f2bf(float f) {
  unsigned int u = __float_as_uint(f);
  if ((u & 0x7F800000u) == 0x7F800000u) return (unsigned short)(u >> 16); // Inf/NaN
  return (unsigned short)((u + 0x7FFFu + ((u >> 16) & 1u)) >> 16);
}

// ---------------------------------------------------------------------------
// f32 -> bf16 bulk convert
// ---------------------------------------------------------------------------
__global__ void cvt_f32_bf16(const float* __restrict__ in,
                             unsigned short* __restrict__ out, long n) {
  long i = (long)blockIdx.x * blockDim.x + threadIdx.x;
  long stride = (long)gridDim.x * blockDim.x;
  for (; i < n; i += stride) out[i] = f2bf(in[i]);
}

// ---------------------------------------------------------------------------
// LayerNorm over D=2048, one row per block (256 threads, 8 elems/thread),
// output bf16 for the following WMMA GEMM.
// ---------------------------------------------------------------------------
__global__ __launch_bounds__(256)
void layernorm_bf16(const float* __restrict__ x, const float* __restrict__ w,
                    const float* __restrict__ bvec,
                    unsigned short* __restrict__ out) {
  __shared__ float red[256];
  const int t = threadIdx.x;
  const size_t row = blockIdx.x;
  const float* xr = x + row * D_MODEL;
  float loc[8];
  float s = 0.f;
#pragma unroll
  for (int i = 0; i < 8; ++i) { loc[i] = xr[t + i * 256]; s += loc[i]; }
  red[t] = s;
  __syncthreads();
  for (int o = 128; o > 0; o >>= 1) {
    if (t < o) red[t] += red[t + o];
    __syncthreads();
  }
  float mu = red[0] * (1.0f / D_MODEL);
  __syncthreads();
  float sq = 0.f;
#pragma unroll
  for (int i = 0; i < 8; ++i) { float d = loc[i] - mu; sq += d * d; }
  red[t] = sq;
  __syncthreads();
  for (int o = 128; o > 0; o >>= 1) {
    if (t < o) red[t] += red[t + o];
    __syncthreads();
  }
  float rstd = rsqrtf(red[0] * (1.0f / D_MODEL) + 1e-5f);
  unsigned short* orow = out + row * D_MODEL;
#pragma unroll
  for (int i = 0; i < 8; ++i) {
    int c = t + i * 256;
    orow[c] = f2bf((loc[i] - mu) * rstd * w[c] + bvec[c]);
  }
}

// ---------------------------------------------------------------------------
// Generic bf16 WMMA GEMM:  out = epilogue(A[MxK] @ W[KxN] + bias[N])
//  EPI==0 : out bf16
//  EPI==1 : out f32 = acc + bias + resid
//  EPI==2 : out bf16 = gelu_tanh(acc + bias)
// Block: 256 threads (8 wave32), 128x128 C tile; wave grid 4x2, each wave
// computes 32x64 (2x4 WMMA 16x16 accumulators). K step 32.
// A tile staged via GLOBAL_LOAD_ASYNC_TO_LDS_B128 when available; W tile is
// transposed element-wise (needs VGPR round trip). Fragment layouts follow
// CDNA5 ISA 7.12.2.
// ---------------------------------------------------------------------------
template <int EPI>
__global__ __launch_bounds__(256)
void gemm_wmma(const unsigned short* __restrict__ A,
               const unsigned short* __restrict__ W,
               const float* __restrict__ bias,
               const float* __restrict__ resid,
               void* __restrict__ outp, int M, int N, int K) {
  __shared__ unsigned short Alds[128][40];
  __shared__ unsigned short Blds[128][40];

  const int tid = threadIdx.x;
  const int lane = tid & 31;
  const int wave = tid >> 5;
  const int mwave = (wave >> 1) * 32;
  const int nwave = (wave & 1) * 64;
  const int mblk = blockIdx.y * 128;
  const int nblk = blockIdx.x * 128;
  const int ml = lane & 15;
  const int lh = lane >> 4;

  v8f acc[2][4] = {};

  for (int kt = 0; kt < K; kt += 32) {
    // stage A tile 128x32
#pragma unroll
    for (int s2 = 0; s2 < 2; ++s2) {
      int c = tid * 2 + s2;
      int row = c >> 2, k8 = (c & 3) * 8;
      const unsigned short* gsrc = A + (size_t)(mblk + row) * K + kt + k8;
#if HAS_ASYNC_LDS
      async_copy_b128(gsrc, &Alds[row][k8]);
#else
      *(uint4*)&Alds[row][k8] = *(const uint4*)gsrc;
#endif
    }
    // stage W tile 32x128, transposed into Blds[n][k]
#pragma unroll
    for (int s2 = 0; s2 < 2; ++s2) {
      int c = tid * 2 + s2;
      int k = c >> 4, n8 = (c & 15) * 8;
      uint4 g = *(const uint4*)(W + (size_t)(kt + k) * N + nblk + n8);
      const unsigned short* hh = (const unsigned short*)&g;
#pragma unroll
      for (int i = 0; i < 8; ++i) Blds[n8 + i][k] = hh[i];
    }
#if HAS_ASYNC_LDS
    wait_async_all();
#endif
    __syncthreads();

    Frag af[2], bf[4];
#pragma unroll
    for (int mi = 0; mi < 2; ++mi) {
      int m = mwave + mi * 16 + ml;
      af[mi].q[0] = *(const uint4*)&Alds[m][lh * 8];
      af[mi].q[1] = *(const uint4*)&Alds[m][16 + lh * 8];
    }
#pragma unroll
    for (int nj = 0; nj < 4; ++nj) {
      int n = nwave + nj * 16 + ml;
      bf[nj].q[0] = *(const uint4*)&Blds[n][lh * 8];
      bf[nj].q[1] = *(const uint4*)&Blds[n][16 + lh * 8];
    }
#pragma unroll
    for (int mi = 0; mi < 2; ++mi)
#pragma unroll
      for (int nj = 0; nj < 4; ++nj)
        acc[mi][nj] = __builtin_amdgcn_wmma_f32_16x16x32_bf16(
            false, af[mi].v, false, bf[nj].v, (short)0, acc[mi][nj], false,
            false);
    __syncthreads();
  }

  // epilogue: C layout -> VGPR r: row = base + r + (lane/16)*8, col = lane%16
#pragma unroll
  for (int mi = 0; mi < 2; ++mi) {
#pragma unroll
    for (int nj = 0; nj < 4; ++nj) {
#pragma unroll
      for (int r = 0; r < 8; ++r) {
        int row = mblk + mwave + mi * 16 + r + lh * 8;
        int col = nblk + nwave + nj * 16 + ml;
        size_t idx = (size_t)row * N + col;
        float v = acc[mi][nj][r] + bias[col];
        if (EPI == 0) {
          ((unsigned short*)outp)[idx] = f2bf(v);
        } else if (EPI == 1) {
          ((float*)outp)[idx] = v + resid[idx];
        } else {
          float u = 0.7978845608028654f * (v + 0.044715f * v * v * v);
          ((unsigned short*)outp)[idx] = f2bf(0.5f * v * (1.0f + tanhf(u)));
        }
      }
    }
  }
}

// ---------------------------------------------------------------------------
// Flash cross-attention: grid (LQ/128, H, B), 256 threads (8 waves).
// Wave w owns 16 query rows. Q fragments stay in registers; K tile key-major
// in LDS (async-staged; K^T B-frags contiguous), V transposed in LDS (P@V
// B-frags contiguous). Online softmax with width-16 shfl reductions matching
// the C matrix lane layout. P goes through wave-private LDS for the C->A
// layout transpose + bf16 convert.
// Dynamic LDS: K 32KB + Vt 32KB + P 32KB = 96KB.
// ---------------------------------------------------------------------------
__global__ __launch_bounds__(256)
void flash_attn(const unsigned short* __restrict__ Q,
                const unsigned short* __restrict__ Kmat,
                const unsigned short* __restrict__ Vmat,
                const unsigned char* __restrict__ mask,
                unsigned short* __restrict__ Oout) {
  extern __shared__ char smem[];
  unsigned short* Klds = (unsigned short*)smem;  // [key 128][d 128]
  unsigned short* Vt = Klds + 128 * 128;         // [d 128][key 128]
  unsigned short* Plds = Vt + 128 * 128;         // [wave 8][q 16][k 128]

  const int tid = threadIdx.x, lane = tid & 31, wave = tid >> 5;
  const int q0 = blockIdx.x * 128;
  const int h = blockIdx.y, b = blockIdx.z;
  const int ml = lane & 15, lh = lane >> 4;

  Frag qf[4];
  {
    const unsigned short* qbase =
        Q + ((size_t)(b * LQ_ + q0 + wave * 16 + ml)) * D_MODEL + h * HEADDIM;
#pragma unroll
    for (int dk = 0; dk < 4; ++dk) {
      qf[dk].q[0] = *(const uint4*)(qbase + dk * 32 + lh * 8);
      qf[dk].q[1] = *(const uint4*)(qbase + dk * 32 + 16 + lh * 8);
    }
  }

  v8f o[8] = {};
  float mrow[8], lrow[8];
#pragma unroll
  for (int r = 0; r < 8; ++r) { mrow[r] = -3.0e38f; lrow[r] = 0.0f; }

  unsigned short* Pw = Plds + wave * 16 * 128;
  const float scale = 0.08838834764831845f;  // 1/sqrt(128)

  for (int kt = 0; kt < LK_ / 128; ++kt) {
    __syncthreads();
#pragma unroll
    for (int i = 0; i < 8; ++i) {
      int c = tid + i * 256;
      int row = c >> 4, d8 = (c & 15) * 8;
      size_t g = (size_t)(b * LK_ + kt * 128 + row) * D_MODEL + h * HEADDIM + d8;
#if HAS_ASYNC_LDS
      async_copy_b128(Kmat + g, &Klds[row * 128 + d8]);
#else
      *(uint4*)&Klds[row * 128 + d8] = *(const uint4*)(Kmat + g);
#endif
      uint4 vg = *(const uint4*)(Vmat + g);
      const unsigned short* hv = (const unsigned short*)&vg;
#pragma unroll
      for (int e = 0; e < 8; ++e) Vt[(d8 + e) * 128 + row] = hv[e];
    }
#if HAS_ASYNC_LDS
    wait_async_all();
#endif
    __syncthreads();

    // S = Q @ K^T  (N dim = keys)
    v8f s[8] = {};
#pragma unroll
    for (int j = 0; j < 8; ++j) {
#pragma unroll
      for (int dk = 0; dk < 4; ++dk) {
        Frag bf;
        int key = j * 16 + ml;
        int dbase = dk * 32 + lh * 8;
        bf.q[0] = *(const uint4*)&Klds[key * 128 + dbase];
        bf.q[1] = *(const uint4*)&Klds[key * 128 + dbase + 16];
        s[j] = __builtin_amdgcn_wmma_f32_16x16x32_bf16(
            false, qf[dk].v, false, bf.v, (short)0, s[j], false, false);
      }
    }

    // scale + mask + running row-max
    float tmax[8];
#pragma unroll
    for (int r = 0; r < 8; ++r) tmax[r] = -3.0e38f;
#pragma unroll
    for (int j = 0; j < 8; ++j) {
      int keyg = kt * 128 + j * 16 + ml;
      bool ok = mask[(size_t)b * LK_ + keyg] != 0;
#pragma unroll
      for (int r = 0; r < 8; ++r) {
        float v = s[j][r] * scale;
        v = ok ? v : -3.0e38f;
        s[j][r] = v;
        tmax[r] = fmaxf(tmax[r], v);
      }
    }
#pragma unroll
    for (int r = 0; r < 8; ++r) {
#pragma unroll
      for (int off = 8; off > 0; off >>= 1)
        tmax[r] = fmaxf(tmax[r], __shfl_xor(tmax[r], off, 16));
      float mn = fmaxf(mrow[r], tmax[r]);
      float alpha = __expf(mrow[r] - mn);
      mrow[r] = mn;
      lrow[r] *= alpha;
#pragma unroll
      for (int t = 0; t < 8; ++t) o[t][r] *= alpha;
    }

    float tsum[8];
#pragma unroll
    for (int r = 0; r < 8; ++r) tsum[r] = 0.0f;
#pragma unroll
    for (int j = 0; j < 8; ++j)
#pragma unroll
      for (int r = 0; r < 8; ++r) {
        float p = __expf(s[j][r] - mrow[r]);
        s[j][r] = p;
        tsum[r] += p;
      }
#pragma unroll
    for (int r = 0; r < 8; ++r) {
#pragma unroll
      for (int off = 8; off > 0; off >>= 1)
        tsum[r] += __shfl_xor(tsum[r], off, 16);
      lrow[r] += tsum[r];
    }

    // C layout -> A layout via wave-private LDS (bf16)
#pragma unroll
    for (int j = 0; j < 8; ++j)
#pragma unroll
      for (int r = 0; r < 8; ++r)
        Pw[(r + lh * 8) * 128 + j * 16 + ml] = f2bf(s[j][r]);

    // O += P @ V
#pragma unroll
    for (int kc = 0; kc < 4; ++kc) {
      Frag pf;
      pf.q[0] = *(const uint4*)&Pw[ml * 128 + kc * 32 + lh * 8];
      pf.q[1] = *(const uint4*)&Pw[ml * 128 + kc * 32 + 16 + lh * 8];
#pragma unroll
      for (int t = 0; t < 8; ++t) {
        Frag vf;
        int d = t * 16 + ml;
        int keybase = kc * 32 + lh * 8;
        vf.q[0] = *(const uint4*)&Vt[d * 128 + keybase];
        vf.q[1] = *(const uint4*)&Vt[d * 128 + keybase + 16];
        o[t] = __builtin_amdgcn_wmma_f32_16x16x32_bf16(
            false, pf.v, false, vf.v, (short)0, o[t], false, false);
      }
    }
  }

  // normalize + store bf16 into attn activation buffer [B*LQ][D]
#pragma unroll
  for (int t = 0; t < 8; ++t)
#pragma unroll
    for (int r = 0; r < 8; ++r) {
      int qrow = q0 + wave * 16 + r + lh * 8;
      int col = h * HEADDIM + t * 16 + ml;
      float v = o[t][r] / fmaxf(lrow[r], 1e-20f);
      Oout[(size_t)(b * LQ_ + qrow) * D_MODEL + col] = f2bf(v);
    }
}

// ---------------------------------------------------------------------------
extern "C" void kernel_launch(void* const* d_in, const int* in_sizes, int n_in,
                              void* d_out, int out_size, void* d_ws,
                              size_t ws_size, hipStream_t stream) {
  (void)in_sizes; (void)n_in; (void)out_size; (void)ws_size;
  const float* hidden = (const float*)d_in[0];
  const float* enc = (const float*)d_in[1];
  const unsigned char* amask = (const unsigned char*)d_in[2];
  const float* ln1w = (const float*)d_in[3];
  const float* ln1b = (const float*)d_in[4];
  const float* q_w = (const float*)d_in[5];
  const float* q_b = (const float*)d_in[6];
  const float* k_w = (const float*)d_in[7];
  const float* k_b = (const float*)d_in[8];
  const float* v_w = (const float*)d_in[9];
  const float* v_b = (const float*)d_in[10];
  const float* cw = (const float*)d_in[11];
  const float* cb = (const float*)d_in[12];
  const float* ln2w = (const float*)d_in[13];
  const float* ln2b = (const float*)d_in[14];
  const float* fcw = (const float*)d_in[15];
  const float* fcb = (const float*)d_in[16];
  const float* pjw = (const float*)d_in[17];
  const float* pjb = (const float*)d_in[18];
  float* out = (float*)d_out;

  const int Mq = BATCH * LQ_;  // 4096
  const int Mk = BATCH * LK_;  // 8192

  char* ws = (char*)d_ws;
  size_t off = 0;
  auto alloc = [&](size_t bytes) -> void* {
    void* p = ws + off;
    off = (off + bytes + 255) & ~(size_t)255;
    return p;
  };

  unsigned short* qw_bf = (unsigned short*)alloc((size_t)D_MODEL * D_MODEL * 2);
  unsigned short* kw_bf = (unsigned short*)alloc((size_t)D_MODEL * D_MODEL * 2);
  unsigned short* vw_bf = (unsigned short*)alloc((size_t)D_MODEL * D_MODEL * 2);
  unsigned short* cw_bf = (unsigned short*)alloc((size_t)D_MODEL * D_MODEL * 2);
  unsigned short* fcw_bf = (unsigned short*)alloc((size_t)D_MODEL * INNER_DIM * 2);
  unsigned short* pjw_bf = (unsigned short*)alloc((size_t)INNER_DIM * D_MODEL * 2);
  unsigned short* x_bf = (unsigned short*)alloc((size_t)Mq * D_MODEL * 2);
  unsigned short* enc_bf = (unsigned short*)alloc((size_t)Mk * D_MODEL * 2);
  unsigned short* q_bf = (unsigned short*)alloc((size_t)Mq * D_MODEL * 2);
  unsigned short* k_bf = (unsigned short*)alloc((size_t)Mk * D_MODEL * 2);
  unsigned short* v_bf = (unsigned short*)alloc((size_t)Mk * D_MODEL * 2);
  unsigned short* attn_bf = (unsigned short*)alloc((size_t)Mq * D_MODEL * 2);
  float* hidden_f = (float*)alloc((size_t)Mq * D_MODEL * 4);
  unsigned short* y_bf = (unsigned short*)alloc((size_t)Mq * D_MODEL * 2);
  unsigned short* h_bf = (unsigned short*)alloc((size_t)Mq * INNER_DIM * 2);

  // weight + activation down-converts
  cvt_f32_bf16<<<2048, 256, 0, stream>>>(q_w, qw_bf, (long)D_MODEL * D_MODEL);
  cvt_f32_bf16<<<2048, 256, 0, stream>>>(k_w, kw_bf, (long)D_MODEL * D_MODEL);
  cvt_f32_bf16<<<2048, 256, 0, stream>>>(v_w, vw_bf, (long)D_MODEL * D_MODEL);
  cvt_f32_bf16<<<2048, 256, 0, stream>>>(cw, cw_bf, (long)D_MODEL * D_MODEL);
  cvt_f32_bf16<<<4096, 256, 0, stream>>>(fcw, fcw_bf, (long)D_MODEL * INNER_DIM);
  cvt_f32_bf16<<<4096, 256, 0, stream>>>(pjw, pjw_bf, (long)INNER_DIM * D_MODEL);
  cvt_f32_bf16<<<4096, 256, 0, stream>>>(enc, enc_bf, (long)Mk * D_MODEL);

  // LN1
  layernorm_bf16<<<Mq, 256, 0, stream>>>(hidden, ln1w, ln1b, x_bf);

  // Q/K/V projections
  gemm_wmma<0><<<dim3(D_MODEL / 128, Mq / 128), 256, 0, stream>>>(
      x_bf, qw_bf, q_b, nullptr, (void*)q_bf, Mq, D_MODEL, D_MODEL);
  gemm_wmma<0><<<dim3(D_MODEL / 128, Mk / 128), 256, 0, stream>>>(
      enc_bf, kw_bf, k_b, nullptr, (void*)k_bf, Mk, D_MODEL, D_MODEL);
  gemm_wmma<0><<<dim3(D_MODEL / 128, Mk / 128), 256, 0, stream>>>(
      enc_bf, vw_bf, v_b, nullptr, (void*)v_bf, Mk, D_MODEL, D_MODEL);

  // attention (96KB dynamic LDS; WGP has 320KB)
  (void)hipFuncSetAttribute((const void*)flash_attn,
                            hipFuncAttributeMaxDynamicSharedMemorySize,
                            96 * 1024);
  flash_attn<<<dim3(LQ_ / 128, NHEAD, BATCH), 256, 96 * 1024, stream>>>(
      q_bf, k_bf, v_bf, amask, attn_bf);

  // cproj + residual -> hidden_f (f32)
  gemm_wmma<1><<<dim3(D_MODEL / 128, Mq / 128), 256, 0, stream>>>(
      attn_bf, cw_bf, cb, hidden, (void*)hidden_f, Mq, D_MODEL, D_MODEL);

  // LN2
  layernorm_bf16<<<Mq, 256, 0, stream>>>(hidden_f, ln2w, ln2b, y_bf);

  // FC + GELU
  gemm_wmma<2><<<dim3(INNER_DIM / 128, Mq / 128), 256, 0, stream>>>(
      y_bf, fcw_bf, fcb, nullptr, (void*)h_bf, Mq, INNER_DIM, D_MODEL);

  // proj + residual -> out (f32)
  gemm_wmma<1><<<dim3(D_MODEL / 128, Mq / 128), 256, 0, stream>>>(
      h_bf, pjw_bf, pjb, hidden_f, (void*)out, Mq, D_MODEL, INNER_DIM);
}
